// Head_16982300689021
// MI455X (gfx1250) — matrix operand order repaired and checked by reading, not compiled, not run on hardware
//
#include <hip/hip_runtime.h>

#define Bq 4
#define Tq 4096
#define Cq 128

typedef __attribute__((ext_vector_type(16))) _Float16 v16h;
typedef __attribute__((ext_vector_type(8)))  _Float16 v8h;
typedef __attribute__((ext_vector_type(8)))  float    v8f;
typedef _Float16 half_t;

#define SHUF16(lo, hi) __builtin_shufflevector(lo, hi, 0, 1, 2, 3, 4, 5, 6, 7, \
                                               8, 9, 10, 11, 12, 13, 14, 15)

// ---------------------------------------------------------------------------
// Kernel 0: convert weights f32 -> f16 AND pre-swizzle into the exact WMMA
// B-fragment layout: wB[mat][nt(8)][ks(4)][lane(32)][j(16)], so the
// projection kernel loads each B operand as one contiguous 32B chunk per lane.
// Softmax scale C^-0.5 folded into Wq.
// ---------------------------------------------------------------------------
__global__ void cvt_weights(const float* __restrict__ Wq,
                            const float* __restrict__ Wk,
                            const float* __restrict__ Wv,
                            half_t* __restrict__ wB) {
  const int idx = blockIdx.x * blockDim.x + threadIdx.x;  // 0 .. 49151
  if (idx >= 3 * 8 * 4 * 32 * 16) return;
  const int j    = idx & 15;
  const int lane = (idx >> 4) & 31;
  const int ks   = (idx >> 9) & 3;
  const int nt   = (idx >> 11) & 7;
  const int mat  = idx >> 14;
  const int krow = ks * 32 + ((lane >> 4) << 4) + j;  // input dim (B K-index)
  const int ncol = nt * 16 + (lane & 15);             // output dim (B N-index)
  const float* W = (mat == 0) ? Wq : (mat == 1) ? Wk : Wv;
  float v = W[krow * Cq + ncol];
  if (mat == 0) v *= 0.08838834764831845f;  // 128^-0.5
  wB[idx] = (half_t)v;
}

// ---------------------------------------------------------------------------
// Kernel 1: fused QKV projection. One WG (8 waves) per 16-row tile of x.
// Each wave computes 3 of 24 16x16 output tiles. x tile staged in LDS f16.
// Outputs: q row-major f16, k row-major f16, v TRANSPOSED f16 (vT[b][dim][t])
// so the attention kernel's B fragments are all per-lane contiguous.
// ---------------------------------------------------------------------------
__global__ __launch_bounds__(256)
void qkv_proj(const float* __restrict__ x, const half_t* __restrict__ wB,
              half_t* __restrict__ q16, half_t* __restrict__ k16,
              half_t* __restrict__ vT) {
  __shared__ half_t xs[16 * Cq];  // 4 KB

  const int tile = blockIdx.x;  // 0 .. B*T/16-1
  const int b  = tile / (Tq / 16);
  const int r0 = (tile % (Tq / 16)) * 16;

  const float* xrow = x + ((size_t)b * Tq + r0) * Cq;
  for (int i = threadIdx.x; i < 16 * Cq; i += 256)
    xs[i] = (half_t)xrow[i];
  __syncthreads();

  const int wave    = threadIdx.x >> 5;
  const int lane    = threadIdx.x & 31;
  const int n       = lane & 15;
  const int halfsel = lane >> 4;
  const int kbA     = halfsel << 4;  // A-fragment K base
  const int mo      = halfsel << 3;  // C-fragment row base

#pragma unroll
  for (int t = 0; t < 3; ++t) {
    const int ot  = wave * 3 + t;  // 0..23
    const int mat = ot >> 3;       // 0=q 1=k 2=v
    const int nt  = ot & 7;
    v8f acc = {};
#pragma unroll
    for (int ks = 0; ks < 4; ++ks) {
      const half_t* ap = xs + n * Cq + ks * 32;
      v16h a = *(const v16h*)(ap + kbA);
      v16h bf = *(const v16h*)(wB +
                 ((((size_t)mat * 8 + nt) * 4 + ks) * 32 + lane) * 16);
      acc = __builtin_amdgcn_wmma_f32_16x16x32_f16(false, a, false, bf,
                                                   (short)0, acc, false, false);
    }
    if (mat < 2) {
      half_t* dst = (mat == 0 ? q16 : k16) +
                    ((size_t)b * Tq + r0) * Cq + nt * 16;
#pragma unroll
      for (int r = 0; r < 8; ++r)
        dst[(size_t)(mo + r) * Cq + n] = (half_t)acc[r];
    } else {
      // vT[b][dim][token]: the 8 row-values per lane are contiguous tokens
      v8h pk;
#pragma unroll
      for (int r = 0; r < 8; ++r) pk[r] = (half_t)acc[r];
      *(v8h*)(vT + ((size_t)b * Cq + nt * 16 + n) * Tq + r0 + mo) = pk;
    }
  }
}

// ---------------------------------------------------------------------------
// Kernel 2: causal flash attention. WG = 4 waves = 64 query rows; each wave
// owns 16 rows. K/V fragments loaded straight from global (L2-resident,
// per-lane contiguous). LDS only used for the P (C->A layout) staging tile.
// ---------------------------------------------------------------------------
#define BR 64
#define BC 32

__global__ __launch_bounds__(128)
void attn(const half_t* __restrict__ q16, const half_t* __restrict__ k16,
          const half_t* __restrict__ vT, float* __restrict__ out) {
  __shared__ half_t ps_[4][16 * BC];  // 4 KB, per-wave P staging

  const int wg   = blockIdx.x;  // B * T/BR
  const int b    = wg / (Tq / BR);
  const int qb   = (wg % (Tq / BR)) * BR;
  const int wave = threadIdx.x >> 5;
  const int lane = threadIdx.x & 31;
  const int qrow = qb + wave * 16;

  const int n       = lane & 15;
  const int halfsel = lane >> 4;
  const int kbA     = halfsel << 4;
  const int kbB     = halfsel << 4;

  // Preload Q fragments (A operand): row-major global, contiguous 16B pairs
  v16h qf[4];
  {
    const half_t* qp = q16 + ((size_t)b * Tq + qrow + n) * Cq;
#pragma unroll
    for (int ks = 0; ks < 4; ++ks) {
      qf[ks] = *(const v16h*)(qp + ks * 32 + kbA);
    }
  }

  v8f   o[8];
  float mrow[8], lrow[8];
#pragma unroll
  for (int i = 0; i < 8; ++i) { o[i] = {}; mrow[i] = -1e30f; lrow[i] = 0.f; }

  const half_t* kb_ptr = k16 + (size_t)b * Tq * Cq;
  const half_t* vb_ptr = vT + (size_t)b * Cq * Tq;

  const int nblk = (qb + BR) / BC;  // uniform across the WG
  for (int j = 0; j < nblk; ++j) {
    const int kg0 = j * BC;

    // Prefetch next key block's fragment lines (global_prefetch_b8)
    if (j + 1 < nblk) {
      __builtin_prefetch(kb_ptr + (size_t)(kg0 + BC + n) * Cq + kbB, 0, 0);
      __builtin_prefetch(vb_ptr + (size_t)n * Tq + kg0 + BC, 0, 0);
    }

    // S = Q * K^T : B fragments are contiguous rows of row-major K
    v8f s0 = {}, s1 = {};
#pragma unroll
    for (int ks = 0; ks < 4; ++ks) {
      v16h b0 = *(const v16h*)(kb_ptr + (size_t)(kg0 + n) * Cq + ks * 32 + kbB);
      v16h b1 = *(const v16h*)(kb_ptr + (size_t)(kg0 + 16 + n) * Cq + ks * 32 + kbB);
      s0 = __builtin_amdgcn_wmma_f32_16x16x32_f16(false, qf[ks], false, b0,
                                                  (short)0, s0, false, false);
      s1 = __builtin_amdgcn_wmma_f32_16x16x32_f16(false, qf[ks], false, b1,
                                                  (short)0, s1, false, false);
    }

    // Causal mask + online softmax (C-fragment row = one VGPR x 16 lanes)
#pragma unroll
    for (int r = 0; r < 8; ++r) {
      const int qg = qrow + (halfsel << 3) + r;
      float a0 = (kg0 + n      <= qg) ? s0[r] : -1e30f;
      float a1 = (kg0 + 16 + n <= qg) ? s1[r] : -1e30f;
      float mx = fmaxf(a0, a1);
#pragma unroll
      for (int off = 8; off; off >>= 1)
        mx = fmaxf(mx, __shfl_xor(mx, off, 32));
      const float mnew = fmaxf(mrow[r], mx);
      const float p0 = __expf(a0 - mnew);
      const float p1 = __expf(a1 - mnew);
      float rs = p0 + p1;
#pragma unroll
      for (int off = 8; off; off >>= 1)
        rs += __shfl_xor(rs, off, 32);
      const float alpha = __expf(mrow[r] - mnew);
      mrow[r] = mnew;
      lrow[r] = lrow[r] * alpha + rs;
#pragma unroll
      for (int nt = 0; nt < 8; ++nt) o[nt][r] *= alpha;
      const int m = (halfsel << 3) + r;
      ps_[wave][(size_t)m * BC + n]      = (half_t)p0;
      ps_[wave][(size_t)m * BC + 16 + n] = (half_t)p1;
    }
    __syncthreads();  // make P visible across lanes (uniform trip count)

    // P fragment (A layout) from LDS: contiguous 16B pairs
    v16h pf;
    {
      const half_t* pp = &ps_[wave][(size_t)n * BC];
      pf = *(const v16h*)(pp + kbA);
    }

    // O += P * V : B fragments are contiguous rows of vT (dim-major)
#pragma unroll
    for (int nt = 0; nt < 8; ++nt) {
      v16h vf = *(const v16h*)(vb_ptr + (size_t)(nt * 16 + n) * Tq + kg0 + kbB);
      o[nt] = __builtin_amdgcn_wmma_f32_16x16x32_f16(false, pf, false, vf,
                                                     (short)0, o[nt], false, false);
    }
  }

  // Normalize and store fp32 output (coalesced across the 16-lane halves)
  float* dst = out + ((size_t)b * Tq + qrow) * Cq;
#pragma unroll
  for (int r = 0; r < 8; ++r) {
    const float inv = 1.0f / lrow[r];
    const int m = (halfsel << 3) + r;
#pragma unroll
    for (int nt = 0; nt < 8; ++nt)
      dst[(size_t)m * Cq + nt * 16 + n] = o[nt][r] * inv;
  }
}

// ---------------------------------------------------------------------------
extern "C" void kernel_launch(void* const* d_in, const int* in_sizes, int n_in,
                              void* d_out, int out_size, void* d_ws, size_t ws_size,
                              hipStream_t stream) {
  const float* x  = (const float*)d_in[0];
  const float* Wq = (const float*)d_in[1];
  const float* Wk = (const float*)d_in[2];
  const float* Wv = (const float*)d_in[3];

  const size_t nQKV = (size_t)Bq * Tq * Cq;              // 2 Mi halves each
  half_t* wB  = (half_t*)d_ws;                           // 96 KB
  half_t* q16 = (half_t*)((char*)d_ws + 3 * Cq * Cq * sizeof(half_t));
  half_t* k16 = q16 + nQKV;
  half_t* vT  = k16 + nQKV;                              // transposed V

  cvt_weights<<<(3 * 8 * 4 * 32 * 16 + 255) / 256, 256, 0, stream>>>(Wq, Wk, Wv, wB);
  qkv_proj<<<Bq * Tq / 16, 256, 0, stream>>>(x, wB, q16, k16, vT);
  attn<<<Bq * Tq / BR, 128, 0, stream>>>(q16, k16, vT, (float*)d_out);
}
